// PoseHeadBlock_53412213293889
// MI455X (gfx1250) — matrix operand order repaired and checked by reading, not compiled, not run on hardware
//
#include <hip/hip_runtime.h>
#include <hip/hip_bf16.h>
#include <math.h>

// ---------- dims ----------
#define Bb   16
#define Jj   16
#define Ll   4096
#define Dd   1024
#define Hh   8
#define HD   128
#define CONDc 512
#define Ss   4
#define BL   (Bb*Ll)      // 65536
#define MH   (Bb*Jj)      // 256

typedef __attribute__((ext_vector_type(16))) __bf16 bf16x16;
typedef __attribute__((ext_vector_type(8)))  float  floatx8;

union ABu { bf16x16 v; uint4 q[2]; };

__device__ __forceinline__ unsigned short f2bf(float f) {
    unsigned int u = __builtin_bit_cast(unsigned int, f);
    unsigned int r = (u + 0x7FFFu + ((u >> 16) & 1u)) >> 16;
    return (unsigned short)r;
}

// A fragment: 16x32 bf16 tile of row-major A[M,lda] at (m0,k0).
// lanes 0-15: row m0+lane, K = k0+0..7 and k0+16..23
// lanes 16-31: row m0+lane-16, K = k0+8..15 and k0+24..31
__device__ __forceinline__ bf16x16 load_a_frag(const unsigned short* __restrict__ A,
                                               size_t lda, int m0, int k0, int ln) {
    int row = m0 + (ln & 15);
    int kb  = k0 + ((ln >> 4) << 3);
    const unsigned short* p = A + (size_t)row * lda + kb;
    ABu u;
    u.q[0] = *reinterpret_cast<const uint4*>(p);
    u.q[1] = *reinterpret_cast<const uint4*>(p + 16);
    return u.v;
}

// B fragment: 32x16 bf16 from Bt stored [N,ldb] (row n = B column n, contiguous over K).
// lanes 0-15: col n0+lane, K = k0+0..7 and k0+16..23 ; lanes 16-31: col n0+lane-16, K = k0+8..15 and k0+24..31
__device__ __forceinline__ bf16x16 load_b_frag(const unsigned short* __restrict__ Bt,
                                               size_t ldb, size_t n0, size_t k0, int ln) {
    size_t col = n0 + (ln & 15);
    size_t kb  = k0 + ((size_t)(ln >> 4) << 3);
    const unsigned short* p = Bt + col * ldb + kb;
    ABu u;
    u.q[0] = *reinterpret_cast<const uint4*>(p);
    u.q[1] = *reinterpret_cast<const uint4*>(p + 16);
    return u.v;
}

__device__ __forceinline__ floatx8 wmma_bf16(bf16x16 a, bf16x16 b, floatx8 c) {
    return __builtin_amdgcn_wmma_f32_16x16x32_bf16(false, a, false, b, (short)0, c, false, false);
}

// ---------- weight transpose-convert: W[K,N] f32 -> Wt[N,K] bf16 ----------
__global__ void convT_k(const float* __restrict__ W, unsigned short* __restrict__ Wt,
                        int K, int N) {
    size_t total = (size_t)K * N;
    for (size_t idx = (size_t)blockIdx.x * 256 + threadIdx.x; idx < total;
         idx += (size_t)gridDim.x * 256) {
        size_t k = idx / N, n = idx % N;
        Wt[n * (size_t)K + k] = f2bf(W[idx]);
    }
}

// ---------- FiLM ----------
__global__ void film_mod_k(const float* __restrict__ cond, const float* __restrict__ fw,
                           const float* __restrict__ fb, float* __restrict__ mod) {
    const int b = blockIdx.x, tid = threadIdx.x;
    __shared__ float cs[CONDc];
    for (int i = tid; i < CONDc; i += 256) cs[i] = cond[b * CONDc + i];
    __syncthreads();
    for (int n = tid; n < 2 * Dd; n += 256) {
        float acc = fb[n];
        for (int k = 0; k < CONDc; ++k) acc += cs[k] * fw[(size_t)k * (2 * Dd) + n];
        mod[b * 2 * Dd + n] = acc;
    }
}

__global__ void film_apply_k(const float* __restrict__ h, const float* __restrict__ mod,
                             float* __restrict__ h1) {
    int idx = blockIdx.x * 256 + threadIdx.x;       // B*J*D = 262144
    int b = idx >> 14;                              // J*D = 16384
    int d = idx & (Dd - 1);
    float sg = mod[b * 2 * Dd + d], sh = mod[b * 2 * Dd + Dd + d];
    h1[idx] = h[idx] * (1.f + sg) + sh;
}

// ---------- LayerNorm -> bf16 ----------
__global__ __launch_bounds__(256) void ln_bf16_k(const float* __restrict__ x,
                                                 const float* __restrict__ g,
                                                 const float* __restrict__ be,
                                                 unsigned short* __restrict__ y, int Dn) {
    const int row = blockIdx.x, tid = threadIdx.x;
    const float* xr = x + (size_t)row * Dn;
    float s = 0.f, ss = 0.f;
    for (int i = tid; i < Dn; i += 256) { float v = xr[i]; s += v; ss += v * v; }
    for (int d2 = 1; d2 < 32; d2 <<= 1) { s += __shfl_xor(s, d2, 32); ss += __shfl_xor(ss, d2, 32); }
    __shared__ float rs[8], rss[8];
    if ((tid & 31) == 0) { rs[tid >> 5] = s; rss[tid >> 5] = ss; }
    __syncthreads();
    float S = 0.f, SS = 0.f;
    for (int w = 0; w < 8; ++w) { S += rs[w]; SS += rss[w]; }
    float mean = S / Dn;
    float var  = SS / Dn - mean * mean;
    float rstd = rsqrtf(var + 1e-5f);
    for (int i = tid; i < Dn; i += 256)
        y[(size_t)row * Dn + i] = f2bf((xr[i] - mean) * rstd * g[i] + be[i]);
}

// ---------- generic WMMA GEMM: C[M,N] = A_bf16[M,K] @ Bt_bf16[N,K]^T + bias ----------
// EPI: 0=f32, 1=f32+residual, 2=bf16, 3=bf16 exact-GELU, 4=bf16 transposed [N,M]
template <int EPI>
__global__ __launch_bounds__(256) void gemm_bf16_k(const unsigned short* __restrict__ A,
                                                   const unsigned short* __restrict__ Bt,
                                                   const float* __restrict__ bias,
                                                   const float* __restrict__ resid,
                                                   void* __restrict__ outp,
                                                   int M, int N, int K) {
    const int tid = threadIdx.x, ln = tid & 31, w = tid >> 5;
    const int wm = w & 3, wn = w >> 2;
    const int m0 = blockIdx.y * 64 + wm * 16;
    const int n0 = blockIdx.x * 128 + wn * 64;
    floatx8 c[4] = {};
    for (int k0 = 0; k0 < K; k0 += 32) {
        if (k0 + 64 < K)
            __builtin_prefetch(A + (size_t)(m0 + (ln & 15)) * K + k0 + 64, 0, 1);
        bf16x16 a = load_a_frag(A, (size_t)K, m0, k0, ln);
#pragma unroll
        for (int j = 0; j < 4; ++j) {
            bf16x16 b = load_b_frag(Bt, (size_t)K, (size_t)(n0 + j * 16), (size_t)k0, ln);
            c[j] = wmma_bf16(a, b, c[j]);
        }
    }
    const int half = ln >> 4, l16 = ln & 15;
#pragma unroll
    for (int j = 0; j < 4; ++j) {
        int ncol = n0 + j * 16 + l16;
        float bb = bias ? bias[ncol] : 0.f;
#pragma unroll
        for (int r = 0; r < 8; ++r) {
            int mrow = m0 + r + 8 * half;
            float v = c[j][r] + bb;
            if constexpr (EPI == 1) v += resid[(size_t)mrow * N + ncol];
            if constexpr (EPI == 0 || EPI == 1)
                ((float*)outp)[(size_t)mrow * N + ncol] = v;
            else if constexpr (EPI == 2)
                ((unsigned short*)outp)[(size_t)mrow * N + ncol] = f2bf(v);
            else if constexpr (EPI == 3) {
                float gl = 0.5f * v * (1.f + erff(v * 0.70710678118654752f));
                ((unsigned short*)outp)[(size_t)mrow * N + ncol] = f2bf(gl);
            } else if constexpr (EPI == 4)
                ((unsigned short*)outp)[(size_t)ncol * M + mrow] = f2bf(v);
        }
    }
}

// ---------- self-attention over J=16 tokens (tiny; VALU) ----------
// K/V tiles staged into LDS with gfx1250 async-to-LDS copies (ASYNCcnt).
__global__ __launch_bounds__(256) void self_attn_k(const float* __restrict__ qkv,
                                                   unsigned short* __restrict__ sa_bf) {
    const int b = blockIdx.x >> 3, hh = blockIdx.x & 7;
    const int tid = threadIdx.x;
    __shared__ float ks[16][128];
    __shared__ float vs[16][128];
    __shared__ float sc[16][17];
    const float scale = 0.088388347648318447f;  // 1/sqrt(128)

    // Async copy K and V head-slices (2 x 8KB) into LDS: each lane moves 16B chunks.
    for (int i = tid; i < 512; i += 256) {           // 512 chunks of 4 floats per array
        int j = i >> 5, d = (i & 31) * 4;
        unsigned ldsk = (unsigned)(size_t)&ks[j][d];
        unsigned ldsv = (unsigned)(size_t)&vs[j][d];
        unsigned long long gk =
            (unsigned long long)(size_t)(qkv + (size_t)(b * 16 + j) * 3072 + hh * 128 + d + 1024);
        unsigned long long gv = gk + 4096ull;        // V is +1024 floats after K
        asm volatile("global_load_async_to_lds_b128 %0, %1, off"
                     :: "v"(ldsk), "v"(gk) : "memory");
        asm volatile("global_load_async_to_lds_b128 %0, %1, off"
                     :: "v"(ldsv), "v"(gv) : "memory");
    }
    asm volatile("s_wait_asynccnt 0x0" ::: "memory");
    __syncthreads();
    {
        int j = tid >> 4, l = tid & 15;
        const float* qrow = qkv + (size_t)(b * 16 + j) * 3072 + hh * 128;
        float acc = 0.f;
        for (int d = 0; d < 128; ++d) acc += qrow[d] * ks[l][d];
        sc[j][l] = acc * scale;
    }
    __syncthreads();
    if (tid < 16) {
        int j = tid;
        float mx = -3.0e38f;
        for (int l = 0; l < 16; ++l) mx = fmaxf(mx, sc[j][l]);
        float sm = 0.f;
        for (int l = 0; l < 16; ++l) { float e = expf(sc[j][l] - mx); sc[j][l] = e; sm += e; }
        float inv = 1.f / sm;
        for (int l = 0; l < 16; ++l) sc[j][l] *= inv;
    }
    __syncthreads();
    for (int i = tid; i < 2048; i += 256) {
        int j = i >> 7, d = i & 127;
        float acc = 0.f;
        for (int l = 0; l < 16; ++l) acc += sc[j][l] * vs[l][d];
        sa_bf[(size_t)(b * 16 + j) * Dd + hh * 128 + d] = f2bf(acc);
    }
}

// ---------- cross-attention: flash per (b,head), WMMA scores + WMMA P@V ----------
__global__ __launch_bounds__(128) void cross_attn_k(
    const unsigned short* __restrict__ qbf,   // [B*J, D] bf16
    const unsigned short* __restrict__ kbf,   // [B*L, D] bf16
    const unsigned short* __restrict__ vT,    // [D, B*L] bf16 (transposed V)
    const unsigned char* __restrict__ tmask,  // [B, L]
    const unsigned char* __restrict__ pam,    // [B, J, L]
    const int* __restrict__ slots,            // [B, L]
    const float* __restrict__ part_bias,      // [H]
    const float* __restrict__ slot_bias,      // [H, S]
    unsigned short* __restrict__ out_bf)      // [B*J, D] bf16
{
    const int bh = blockIdx.x, b = bh / Hh, hh = bh % Hh;
    const int tid = threadIdx.x, w = tid >> 5, ln = tid & 31;
    const int half = ln >> 4, l16 = ln & 15;

    __shared__ unsigned short Pt[4][16][40];  // per-wave P tile (16 j x 32 l), pitch 40
    __shared__ float Ocomb[4][16][128];
    __shared__ float mComb[4][16];
    __shared__ float lComb[4][16];

    bf16x16 qa[4];
#pragma unroll
    for (int f = 0; f < 4; ++f)
        qa[f] = load_a_frag(qbf, (size_t)Dd, b * Jj, hh * HD + f * 32, ln);

    floatx8 O[8] = {};
    float m8[8], s8[8];
#pragma unroll
    for (int r = 0; r < 8; ++r) { m8[r] = -3.0e38f; s8[r] = 0.f; }

    const float scale = 0.088388347648318447f;
    const float pbh = part_bias[hh];
    const unsigned char* pamb = pam + (size_t)(b * Jj) * Ll;

    for (int it = 0; it < 32; ++it) {
        const int l0 = w * 1024 + it * 32;
        floatx8 S0 = {}, S1 = {};
#pragma unroll
        for (int f = 0; f < 4; ++f) {
            bf16x16 kb0 = load_b_frag(kbf, (size_t)Dd, (size_t)(b * Ll + l0),      (size_t)(hh * HD + f * 32), ln);
            bf16x16 kb1 = load_b_frag(kbf, (size_t)Dd, (size_t)(b * Ll + l0 + 16), (size_t)(hh * HD + f * 32), ln);
            S0 = wmma_bf16(qa[f], kb0, S0);
            S1 = wmma_bf16(qa[f], kb1, S1);
        }
        const int c0 = l0 + l16, c1 = c0 + 16;
        const bool mk0 = tmask[b * Ll + c0] != 0;
        const bool mk1 = tmask[b * Ll + c1] != 0;
        int sl0 = slots[b * Ll + c0]; sl0 = sl0 < 0 ? 0 : (sl0 > Ss - 1 ? Ss - 1 : sl0);
        int sl1 = slots[b * Ll + c1]; sl1 = sl1 < 0 ? 0 : (sl1 > Ss - 1 ? Ss - 1 : sl1);
        const float sb0 = slot_bias[hh * Ss + sl0];
        const float sb1 = slot_bias[hh * Ss + sl1];

        float p0v[8], p1v[8];
#pragma unroll
        for (int r = 0; r < 8; ++r) {
            const int j = r + 8 * half;
            float pb0 = pamb[(size_t)j * Ll + c0] ? pbh : 0.f;
            float pb1 = pamb[(size_t)j * Ll + c1] ? pbh : 0.f;
            float s0 = mk0 ? (S0[r] * scale + pb0 + sb0) : -3.0e38f;
            float s1 = mk1 ? (S1[r] * scale + pb1 + sb1) : -3.0e38f;
            float tm = fmaxf(s0, s1);
            for (int d2 = 1; d2 < 16; d2 <<= 1) tm = fmaxf(tm, __shfl_xor(tm, d2, 32));
            float mnew = fmaxf(m8[r], tm);
            float scr = expf(m8[r] - mnew);
            float e0 = expf(s0 - mnew), e1 = expf(s1 - mnew);
            float rssum = e0 + e1;
            for (int d2 = 1; d2 < 16; d2 <<= 1) rssum += __shfl_xor(rssum, d2, 32);
            s8[r] = s8[r] * scr + rssum;
            m8[r] = mnew;
#pragma unroll
            for (int f = 0; f < 8; ++f) O[f][r] *= scr;
            p0v[r] = e0; p1v[r] = e1;
        }
#pragma unroll
        for (int r = 0; r < 8; ++r) {
            const int j = r + 8 * half;
            Pt[w][j][l16]      = f2bf(p0v[r]);
            Pt[w][j][16 + l16] = f2bf(p1v[r]);
        }
        __syncthreads();
        // P as A-fragment (16x32) from LDS
        const unsigned short* pp = &Pt[w][l16][half * 8];
        ABu u;
        u.q[0] = *reinterpret_cast<const uint4*>(pp);
        u.q[1] = *reinterpret_cast<const uint4*>(pp + 16);
        bf16x16 pa = u.v;
#pragma unroll
        for (int f = 0; f < 8; ++f) {
            bf16x16 vb = load_b_frag(vT, (size_t)BL, (size_t)(hh * HD + f * 16),
                                     (size_t)(b * Ll + l0), ln);
            O[f] = wmma_bf16(pa, vb, O[f]);
        }
        __syncthreads();
    }

#pragma unroll
    for (int f = 0; f < 8; ++f)
#pragma unroll
        for (int r = 0; r < 8; ++r)
            Ocomb[w][r + 8 * half][f * 16 + l16] = O[f][r];
    if (l16 == 0) {
#pragma unroll
        for (int r = 0; r < 8; ++r) {
            mComb[w][r + 8 * half] = m8[r];
            lComb[w][r + 8 * half] = s8[r];
        }
    }
    __syncthreads();
    for (int i = tid; i < Jj * HD; i += 128) {
        int j = i >> 7, d = i & 127;
        float Mx = -3.0e38f;
        for (int ww = 0; ww < 4; ++ww) Mx = fmaxf(Mx, mComb[ww][j]);
        float Ssum = 0.f, num = 0.f;
        for (int ww = 0; ww < 4; ++ww) {
            float e = expf(mComb[ww][j] - Mx);
            Ssum += e * lComb[ww][j];
            num  += e * Ocomb[ww][j][d];
        }
        float val = num / fmaxf(Ssum, 1e-30f);
        out_bf[(size_t)(b * Jj + j) * Dd + hh * HD + d] = f2bf(val);
    }
}

// ---------- host orchestration ----------
extern "C" void kernel_launch(void* const* d_in, const int* in_sizes, int n_in,
                              void* d_out, int out_size, void* d_ws, size_t ws_size,
                              hipStream_t stream) {
    const float* h        = (const float*)d_in[0];
    const float* cond     = (const float*)d_in[1];
    const float* token_kv = (const float*)d_in[2];
    const unsigned char* tmask = (const unsigned char*)d_in[3];
    const unsigned char* pam   = (const unsigned char*)d_in[4];
    const int*   slots    = (const int*)d_in[5];
    const float* film_w = (const float*)d_in[6];
    const float* film_b = (const float*)d_in[7];
    const float* sa_g = (const float*)d_in[8],  *sa_b = (const float*)d_in[9];
    const float* ca_g = (const float*)d_in[10], *ca_b = (const float*)d_in[11];
    const float* kv_g = (const float*)d_in[12], *kv_b = (const float*)d_in[13];
    const float* ff_g = (const float*)d_in[14], *ff_b = (const float*)d_in[15];
    const float* in_w = (const float*)d_in[16], *in_b = (const float*)d_in[17];
    const float* ao_w = (const float*)d_in[18], *ao_b = (const float*)d_in[19];
    const float* qw = (const float*)d_in[20], *qb = (const float*)d_in[21];
    const float* kw = (const float*)d_in[22], *kb = (const float*)d_in[23];
    const float* vw = (const float*)d_in[24], *vb = (const float*)d_in[25];
    const float* ow = (const float*)d_in[26], *ob = (const float*)d_in[27];
    const float* part_bias = (const float*)d_in[28];
    const float* slot_bias = (const float*)d_in[29];
    const float* w1 = (const float*)d_in[30], *b1 = (const float*)d_in[31];
    const float* w2 = (const float*)d_in[32], *b2 = (const float*)d_in[33];

    char* wsb = (char*)d_ws;
    size_t off = 0;
    auto alloc = [&](size_t bytes) {
        size_t cur = off;
        off = (off + bytes + 255) & ~(size_t)255;
        return wsb + cur;
    };
    unsigned short* WinT = (unsigned short*)alloc((size_t)3072 * 1024 * 2);
    unsigned short* WaoT = (unsigned short*)alloc((size_t)1024 * 1024 * 2);
    unsigned short* WqT  = (unsigned short*)alloc((size_t)1024 * 1024 * 2);
    unsigned short* WkT  = (unsigned short*)alloc((size_t)1024 * 1024 * 2);
    unsigned short* WvT  = (unsigned short*)alloc((size_t)1024 * 1024 * 2);
    unsigned short* WoT  = (unsigned short*)alloc((size_t)1024 * 1024 * 2);
    unsigned short* W1T  = (unsigned short*)alloc((size_t)4096 * 1024 * 2);
    unsigned short* W2T  = (unsigned short*)alloc((size_t)1024 * 4096 * 2);
    float*          modb = (float*)alloc((size_t)Bb * 2 * Dd * 4);
    float*          h1   = (float*)alloc((size_t)MH * Dd * 4);
    unsigned short* h2bf = (unsigned short*)alloc((size_t)MH * Dd * 2);
    float*          qkv  = (float*)alloc((size_t)MH * 3 * Dd * 4);
    unsigned short* sabf = (unsigned short*)alloc((size_t)MH * Dd * 2);
    float*          hA   = (float*)alloc((size_t)MH * Dd * 4);
    unsigned short* h3bf = (unsigned short*)alloc((size_t)MH * Dd * 2);
    unsigned short* kvnb = (unsigned short*)alloc((size_t)BL * Dd * 2);
    unsigned short* qbf  = (unsigned short*)alloc((size_t)MH * Dd * 2);
    unsigned short* kbfw = (unsigned short*)alloc((size_t)BL * Dd * 2);
    unsigned short* vTbf = (unsigned short*)alloc((size_t)Dd * BL * 2);
    unsigned short* cabf = (unsigned short*)alloc((size_t)MH * Dd * 2);
    float*          hB   = (float*)alloc((size_t)MH * Dd * 4);
    unsigned short* xnbf = (unsigned short*)alloc((size_t)MH * Dd * 2);
    unsigned short* gbf  = (unsigned short*)alloc((size_t)MH * 4 * Dd * 2);
    (void)ws_size; (void)in_sizes; (void)n_in; (void)out_size;

    // 1) convert weights to bf16 [N,K]
    convT_k<<<2048, 256, 0, stream>>>(in_w, WinT, 1024, 3072);
    convT_k<<<2048, 256, 0, stream>>>(ao_w, WaoT, 1024, 1024);
    convT_k<<<2048, 256, 0, stream>>>(qw,   WqT,  1024, 1024);
    convT_k<<<2048, 256, 0, stream>>>(kw,   WkT,  1024, 1024);
    convT_k<<<2048, 256, 0, stream>>>(vw,   WvT,  1024, 1024);
    convT_k<<<2048, 256, 0, stream>>>(ow,   WoT,  1024, 1024);
    convT_k<<<4096, 256, 0, stream>>>(w1,   W1T,  1024, 4096);
    convT_k<<<4096, 256, 0, stream>>>(w2,   W2T,  4096, 1024);

    // 2) FiLM
    film_mod_k<<<Bb, 256, 0, stream>>>(cond, film_w, film_b, modb);
    film_apply_k<<<(MH * Dd) / 256, 256, 0, stream>>>(h, modb, h1);

    // 3) self-attention
    ln_bf16_k<<<MH, 256, 0, stream>>>(h1, sa_g, sa_b, h2bf, Dd);
    gemm_bf16_k<0><<<dim3(3072 / 128, MH / 64), 256, 0, stream>>>(h2bf, WinT, in_b, nullptr,
                                                                  qkv, MH, 3072, 1024);
    self_attn_k<<<Bb * Hh, 256, 0, stream>>>(qkv, sabf);
    gemm_bf16_k<1><<<dim3(1024 / 128, MH / 64), 256, 0, stream>>>(sabf, WaoT, ao_b, h1,
                                                                  hA, MH, 1024, 1024);

    // 4) cross-attention
    ln_bf16_k<<<MH, 256, 0, stream>>>(hA, ca_g, ca_b, h3bf, Dd);
    ln_bf16_k<<<BL, 256, 0, stream>>>(token_kv, kv_g, kv_b, kvnb, Dd);
    gemm_bf16_k<2><<<dim3(1024 / 128, MH / 64), 256, 0, stream>>>(h3bf, WqT, qb, nullptr,
                                                                  qbf, MH, 1024, 1024);
    gemm_bf16_k<2><<<dim3(1024 / 128, BL / 64), 256, 0, stream>>>(kvnb, WkT, kb, nullptr,
                                                                  kbfw, BL, 1024, 1024);
    gemm_bf16_k<4><<<dim3(1024 / 128, BL / 64), 256, 0, stream>>>(kvnb, WvT, vb, nullptr,
                                                                  vTbf, BL, 1024, 1024);
    cross_attn_k<<<Bb * Hh, 128, 0, stream>>>(qbf, kbfw, vTbf, tmask, pam, slots,
                                              part_bias, slot_bias, cabf);
    gemm_bf16_k<1><<<dim3(1024 / 128, MH / 64), 256, 0, stream>>>(cabf, WoT, ob, hA,
                                                                  hB, MH, 1024, 1024);

    // 5) FFN
    ln_bf16_k<<<MH, 256, 0, stream>>>(hB, ff_g, ff_b, xnbf, Dd);
    gemm_bf16_k<3><<<dim3(4096 / 128, MH / 64), 256, 0, stream>>>(xnbf, W1T, b1, nullptr,
                                                                  gbf, MH, 4096, 1024);
    gemm_bf16_k<1><<<dim3(1024 / 128, MH / 64), 256, 0, stream>>>(gbf, W2T, b2, hB,
                                                                  (float*)d_out, MH, 1024, 4096);
}
